// AttnRNNDecoder_13288628814555
// MI455X (gfx1250) — matrix operand order repaired and checked by reading, not compiled, not run on hardware
//
#include <hip/hip_runtime.h>

// ---------------------------------------------------------------------------
// CDNA5 (gfx1250) persistent attention-LSTM decoder.
// One launch runs all 127 timesteps; phases are separated by a software
// grid barrier (64 workgroups x 8 waves, expected co-resident).
// All GEMMs run on v_wmma_f32_16x16x32_f16 with f32 accumulation; weights are
// converted to f16 once at kernel start into the workspace (L2-resident).
// The gates-GEMM weight slab (2 x 128KB per WG) is staged into LDS ONCE via
// the Tensor Data Mover (TENSOR_LOAD_TO_LDS + s_wait_tensorcnt) and re-read
// from LDS (ds_load) on every timestep.
// ---------------------------------------------------------------------------

typedef __attribute__((ext_vector_type(16))) _Float16 v16h;
typedef __attribute__((ext_vector_type(8)))  float    v8f;
typedef __attribute__((ext_vector_type(4)))  unsigned u32x4;
typedef __attribute__((ext_vector_type(8)))  int      i32x8;
typedef __attribute__((ext_vector_type(4)))  int      i32x4;

#define B_    64
#define L_    512
#define DE_   512
#define DH_   512
#define DO_   256
#define VB_   100
#define CV_   1000
#define T_    128
#define KATT  768            // DO_ + DH_
#define NG    2048           // 4*DH_
#define NV_   1100           // CV_ + VB_
#define PLD   1104           // padded proj row
#define CN    1024           // common vocab padded to WMMA tiles

#define NWG   64
#define NTHR  256
#define WPW   8              // waves per workgroup
#define NWAVES (NWG*WPW)     // 512
#define SLAB  (128*DH_)      // 128 weight rows x 512 K, halves (128KB bytes)

#define NEG_INF (-__builtin_inff())

// -------------------- workspace layout (all offsets 256B aligned) ----------
static constexpr size_t OFF_H     = 0;
static constexpr size_t OFF_C     = OFF_H     + (size_t)B_*DH_*4;
static constexpr size_t OFF_AW    = OFF_C     + (size_t)B_*DH_*4;
static constexpr size_t OFF_CTX   = OFF_AW    + (size_t)B_*L_*4;
static constexpr size_t OFF_GATES = OFF_CTX   + (size_t)B_*DE_*4;
static constexpr size_t OFF_PROJ  = OFF_GATES + (size_t)B_*NG*4;
static constexpr size_t OFF_BAR   = OFF_PROJ  + (size_t)B_*PLD*4;
static constexpr size_t OFF_WATT  = OFF_BAR   + 256;
static constexpr size_t OFF_WCMB  = OFF_WATT  + (size_t)L_*KATT*2;
static constexpr size_t OFF_WIH   = OFF_WCMB  + (size_t)DH_*KATT*2;
static constexpr size_t OFF_WHH   = OFF_WIH   + (size_t)NG*DH_*2;
static constexpr size_t OFF_WOUT  = OFF_WHH   + (size_t)NG*DH_*2;
static constexpr size_t OFF_CEMB  = OFF_WOUT  + (size_t)DO_*DH_*2;
static constexpr size_t OFF_XH    = OFF_CEMB  + (size_t)CN*DO_*2;
static constexpr size_t OFF_XC    = OFF_XH    + (size_t)B_*KATT*2;
static constexpr size_t OFF_R16   = OFF_XC    + (size_t)B_*KATT*2;
static constexpr size_t OFF_H16   = OFF_R16   + (size_t)B_*DH_*2;
static constexpr size_t OFF_O16   = OFF_H16   + (size_t)B_*DH_*2;
static constexpr size_t WS_TOTAL  = OFF_O16   + (size_t)B_*DO_*2;

// -------------------- WMMA fragment helpers (wave32 layouts) ---------------
// A fragment: 16x32 f16, row-major source [M x K], lda = K (ISA 7.12.2).
__device__ __forceinline__ v16h frag_a(const _Float16* A, int lda, int m0,
                                       int k0, int lane) {
  const int m  = m0 + (lane & 15);
  const int kh = ((lane >> 4) & 1) << 3;       // 0 or 8
  const _Float16* p = A + (size_t)m * lda + k0;
  v16h a;
#pragma unroll
  for (int v = 0; v < 4; ++v) { a[2*v] = p[kh + 2*v];          a[2*v+1] = p[kh + 2*v + 1]; }
#pragma unroll
  for (int v = 4; v < 8; ++v) { a[2*v] = p[16 + kh + 2*(v-4)]; a[2*v+1] = p[16 + kh + 2*(v-4) + 1]; }
  return a;
}

// B fragment: 32x16 f16 from weight stored [N x K] row-major (B(k,n)=W[n,k]).
// lanes 0-15 hold K=0..7,16..23, lanes 16-31 hold K=8..15,24..31 (matches frag_a).
__device__ __forceinline__ v16h frag_b(const _Float16* W, int ldw, int n0,
                                       int k0, int lane) {
  const int n  = n0 + (lane & 15);
  const int kb = k0 + (((lane >> 4) & 1) << 3);
  const _Float16* p = W + (size_t)n * ldw + kb;
  v16h b;
#pragma unroll
  for (int i = 0; i < 16; ++i) b[i] = p[i < 8 ? i : i + 8];
  return b;
}

__device__ __forceinline__ v8f wmma_kloop(const _Float16* A, int lda,
                                          const _Float16* Wnk, int ldw,
                                          int m0, int n0, int K, int lane, v8f acc) {
  for (int k0 = 0; k0 < K; k0 += 32) {
    v16h a = frag_a(A, lda, m0, k0, lane);
    v16h b = frag_b(Wnk, ldw, n0, k0, lane);
    acc = __builtin_amdgcn_wmma_f32_16x16x32_f16(false, a, false, b,
                                                 (short)0, acc, false, false);
  }
  return acc;
}

// C/D layout: n = n0 + lane%16, m = m0 + vgprRow + 8*(lane/16)
__device__ __forceinline__ void store_c_f32(float* C, int ldc, int m0, int n0,
                                            int lane, v8f acc) {
  const int n  = n0 + (lane & 15);
  const int mb = m0 + (((lane >> 4) & 1) << 3);
#pragma unroll
  for (int r = 0; r < 8; ++r) C[(size_t)(mb + r) * ldc + n] = acc[r];
}

// -------------------- block reductions (LDS) -------------------------------
__device__ __forceinline__ float block_max(float v, float* s) {
#pragma unroll
  for (int o = 16; o > 0; o >>= 1) v = fmaxf(v, __shfl_xor(v, o, 32));
  const int w = threadIdx.x >> 5;
  if ((threadIdx.x & 31) == 0) s[w] = v;
  __syncthreads();
  float r = s[0];
#pragma unroll
  for (int i = 1; i < WPW; ++i) r = fmaxf(r, s[i]);
  __syncthreads();
  return r;
}

__device__ __forceinline__ float block_sum(float v, float* s) {
#pragma unroll
  for (int o = 16; o > 0; o >>= 1) v += __shfl_xor(v, o, 32);
  const int w = threadIdx.x >> 5;
  if ((threadIdx.x & 31) == 0) s[w] = v;
  __syncthreads();
  float r = s[0];
#pragma unroll
  for (int i = 1; i < WPW; ++i) r += s[i];
  __syncthreads();
  return r;
}

// -------------------- device-wide barrier ----------------------------------
__device__ __forceinline__ void grid_barrier(unsigned* cnt, unsigned* gen,
                                             unsigned& myGen) {
  __syncthreads();
  if (threadIdx.x == 0) {
    __threadfence();
    if (atomicAdd(cnt, 1u) == NWG - 1u) {
      atomicExch(cnt, 0u);
      atomicAdd(gen, 1u);
    } else {
      while (atomicAdd(gen, 0u) <= myGen) __builtin_amdgcn_s_sleep(1);
    }
    __threadfence();
  }
  myGen++;
  __syncthreads();
}

__device__ __forceinline__ float sigf(float x) { return 1.0f / (1.0f + __expf(-x)); }

// ---- Tensor Data Mover: load a [rows x cols] f16 tile (row-major, leading
// dim srcLd elements) from global into LDS at byte offset ldsOff.
// D# bitfields per CDNA5 ISA ch.8: group0 {count=1, lds_addr, global_addr,
// type=2}; group1 {data_size=1(2B), tensor_dim0/1, tile_dim0/1, dim0_stride}.
// This toolchain exposes the 6-arg builtin: (g0, g1, g2, g3, g4, cpol);
// groups 2/3/4 are unused for a 2-D tile and passed as zeros.
#if __has_builtin(__builtin_amdgcn_tensor_load_to_lds)
#define HAVE_TDM 1
__device__ __forceinline__ void tdm_load_2d_f16(const _Float16* gsrc,
                                                unsigned ldsOff, unsigned rows,
                                                unsigned cols, unsigned srcLd) {
  unsigned long long ga = (unsigned long long)(size_t)gsrc;
  u32x4 g0;
  g0[0] = 1u;                                         // count=1 (valid user D#)
  g0[1] = ldsOff;                                     // lds_addr
  g0[2] = (unsigned)(ga & 0xffffffffu);               // global_addr[31:0]
  g0[3] = (unsigned)((ga >> 32) & 0x01ffffffu)        // global_addr[56:32]
        | (2u << 30);                                 // type=2 ("image")
  i32x8 g1;
  g1[0] = (int)(1u << 16);                            // data_size=1 -> 2 bytes
  g1[1] = (int)((cols & 0xffffu) << 16);              // tensor_dim0[15:0]
  g1[2] = (int)(((cols >> 16) & 0xffffu)              // tensor_dim0[31:16]
        | ((rows & 0xffffu) << 16));                  // tensor_dim1[15:0]
  g1[3] = (int)(((rows >> 16) & 0xffffu)              // tensor_dim1[31:16]
        | ((cols & 0xffffu) << 16));                  // tile_dim0 = cols
  g1[4] = (int)(rows & 0xffffu);                      // tile_dim1 = rows
  g1[5] = (int)srcLd;                                 // tensor_dim0_stride[31:0]
  g1[6] = 0;
  g1[7] = 0;
  i32x4 gz4; gz4[0] = 0; gz4[1] = 0; gz4[2] = 0; gz4[3] = 0;  // groups 2/3 (2D: unused)
  i32x8 gz8;
#pragma unroll
  for (int i = 0; i < 8; ++i) gz8[i] = 0;
  __builtin_amdgcn_tensor_load_to_lds(g0, g1, gz4, gz4, gz8, 0);
}
#else
#define HAVE_TDM 0
#endif

// ---- embedding gather + pack [x_t , h] in f16 (one batch row per block) ---
__device__ __forceinline__ void pack_step_inputs(int b, int t,
    const int* __restrict__ tgt, const float* __restrict__ cemb,
    const float* __restrict__ obe, const float* __restrict__ hbuf,
    _Float16* __restrict__ xh16, _Float16* __restrict__ xc16) {
  const int idx = tgt[b * T_ + t];
  const int d   = threadIdx.x;                        // NTHR == DO_
  float xv;
  if (idx < CV_) {
    int ci = idx < 0 ? 0 : (idx > CV_ - 1 ? CV_ - 1 : idx);
    xv = cemb[(size_t)ci * DO_ + d];
  } else {
    int bi = idx - CV_; bi = bi > VB_ - 1 ? VB_ - 1 : bi;
    xv = obe[((size_t)b * VB_ + bi) * DO_ + d];
  }
  const _Float16 xh = (_Float16)xv;
  xh16[(size_t)b * KATT + d] = xh;
  xc16[(size_t)b * KATT + d] = xh;
  // h16 itself is maintained by the LSTM phase (zero-init covers t=0).
  for (int j = threadIdx.x; j < DH_; j += NTHR)
    xh16[(size_t)b * KATT + DO_ + j] = (_Float16)hbuf[(size_t)b * DH_ + j];
}

// ---------------------------------------------------------------------------
__global__ __launch_bounds__(NTHR)
void attn_lstm_decoder(const float* __restrict__ enc,          // [B,L,DE]
                       const unsigned char* __restrict__ encM, // [B,L]
                       const float* __restrict__ obe,          // [B,VB,DO]
                       const unsigned char* __restrict__ obeM, // [B,VB]
                       const int* __restrict__ tgt,            // [B,T]
                       const float* __restrict__ cemb,         // [CV,DO]
                       const float* __restrict__ Wattn, const float* __restrict__ battn,
                       const float* __restrict__ Wcomb, const float* __restrict__ bcomb,
                       const float* __restrict__ Wih,   const float* __restrict__ bih,
                       const float* __restrict__ Whh,   const float* __restrict__ bhh,
                       const float* __restrict__ Wout,  const float* __restrict__ bout,
                       float* __restrict__ dout,               // [B,T-1,NV]
                       unsigned char* __restrict__ ws) {
  __shared__ float sred[WPW];
  extern __shared__ _Float16 wlds[];   // [2*SLAB]: Wih slab | Whh slab (256KB)

  float*     hbuf  = (float*)(ws + OFF_H);
  float*     cbuf  = (float*)(ws + OFF_C);
  float*     aw    = (float*)(ws + OFF_AW);
  float*     ctxb  = (float*)(ws + OFF_CTX);
  float*     gates = (float*)(ws + OFF_GATES);
  float*     proj  = (float*)(ws + OFF_PROJ);
  unsigned*  bcnt  = (unsigned*)(ws + OFF_BAR);
  unsigned*  bgen  = bcnt + 1;
  _Float16*  wat16 = (_Float16*)(ws + OFF_WATT);
  _Float16*  wcb16 = (_Float16*)(ws + OFF_WCMB);
  _Float16*  wih16 = (_Float16*)(ws + OFF_WIH);
  _Float16*  whh16 = (_Float16*)(ws + OFF_WHH);
  _Float16*  wo16  = (_Float16*)(ws + OFF_WOUT);
  _Float16*  ce16  = (_Float16*)(ws + OFF_CEMB);
  _Float16*  xh16  = (_Float16*)(ws + OFF_XH);
  _Float16*  xc16  = (_Float16*)(ws + OFF_XC);
  _Float16*  r16   = (_Float16*)(ws + OFF_R16);
  _Float16*  h16   = (_Float16*)(ws + OFF_H16);
  _Float16*  o16   = (_Float16*)(ws + OFF_O16);

  const int gtid = blockIdx.x * NTHR + threadIdx.x;
  const int gthr = NWG * NTHR;
  const int lane = threadIdx.x & 31;
  const int wid  = gtid >> 5;
  const int b    = blockIdx.x;         // NWG == B_
  unsigned  myGen = 0;

  // ---- P0: one-time f32 -> f16 weight conversion (then L2-resident) ------
  for (int i = gtid; i < L_*KATT;  i += gthr) wat16[i] = (_Float16)Wattn[i];
  for (int i = gtid; i < DH_*KATT; i += gthr) wcb16[i] = (_Float16)Wcomb[i];
  for (int i = gtid; i < NG*DH_;   i += gthr) wih16[i] = (_Float16)Wih[i];
  for (int i = gtid; i < NG*DH_;   i += gthr) whh16[i] = (_Float16)Whh[i];
  for (int i = gtid; i < DO_*DH_;  i += gthr) wo16[i]  = (_Float16)Wout[i];
  for (int i = gtid; i < CV_*DO_;  i += gthr) ce16[i]  = (_Float16)cemb[i];
  // rows CV_..CN-1 of ce16 stay 0 from the host-side memset.
  grid_barrier(bcnt, bgen, myGen);

  // ---- P1: stage this WG's gates-weight slab into LDS once (TDM) ---------
  // WG k always computes gate columns [(k%16)*128, +128): tile == wid below.
  {
    const int rr = blockIdx.x & 15;
    const _Float16* gih = wih16 + (size_t)rr * 128 * DH_;
    const _Float16* ghh = whh16 + (size_t)rr * 128 * DH_;
#if HAVE_TDM
    if (threadIdx.x < 32) {            // one wave issues both DMAs
      const unsigned lds0 =
          (unsigned)((size_t)(void*)&wlds[0] & 0xffffffffu); // LDS byte offset
      tdm_load_2d_f16(gih, lds0,               128u, (unsigned)DH_, (unsigned)DH_);
      tdm_load_2d_f16(ghh, lds0 + SLAB*2u,     128u, (unsigned)DH_, (unsigned)DH_);
      __builtin_amdgcn_s_wait_tensorcnt(0);
    }
    __syncthreads();
#else
    for (int i = threadIdx.x; i < SLAB; i += NTHR) {
      wlds[i]        = gih[i];
      wlds[SLAB + i] = ghh[i];
    }
    __syncthreads();
#endif
  }

  // ---- prologue: pack step-0 inputs --------------------------------------
  pack_step_inputs(b, 0, tgt, cemb, obe, hbuf, xh16, xc16);
  grid_barrier(bcnt, bgen, myGen);

  for (int t = 0; t < T_ - 1; ++t) {
    // ---- A1: attention logits  aw = [x,h] @ Wattn^T + b_attn (+mask) -----
    {
      const int NT = (B_/16) * (L_/16);             // 128 tiles
      for (int tile = wid; tile < NT; tile += NWAVES) {
        const int mt = tile / (L_/16), nt = tile % (L_/16);
        const int n  = nt*16 + (lane & 15);
        v8f acc; const float bi = battn[n];
#pragma unroll
        for (int r = 0; r < 8; ++r) acc[r] = bi;
        acc = wmma_kloop(xh16, KATT, wat16, KATT, mt*16, nt*16, KATT, lane, acc);
        const int mb = mt*16 + (((lane >> 4) & 1) << 3);
#pragma unroll
        for (int r = 0; r < 8; ++r) {
          const int m = mb + r;
          float v = acc[r];
          if (!encM[(size_t)m * L_ + n]) v = NEG_INF;
          aw[(size_t)m * L_ + n] = v;
        }
      }
    }
    grid_barrier(bcnt, bgen, myGen);

    // ---- A2/A3: softmax over L + context matvec (block = batch row) ------
    {
      float lm = NEG_INF;
      for (int l = threadIdx.x; l < L_; l += NTHR) lm = fmaxf(lm, aw[(size_t)b*L_ + l]);
      const float mx = block_max(lm, sred);
      float ls = 0.0f;
      for (int l = threadIdx.x; l < L_; l += NTHR) {
        float e = __expf(aw[(size_t)b*L_ + l] - mx);
        aw[(size_t)b*L_ + l] = e; ls += e;
      }
      const float inv = 1.0f / block_sum(ls, sred);
      for (int l = threadIdx.x; l < L_; l += NTHR) aw[(size_t)b*L_ + l] *= inv;
      __syncthreads();
      for (int d = threadIdx.x; d < DE_; d += NTHR) {
        float acc = 0.0f;
        const float* er = enc + (size_t)b * L_ * DE_ + d;
        for (int l = 0; l < L_; ++l) {
          if ((l & 15) == 0 && l + 16 < L_)
            __builtin_prefetch(er + (size_t)(l + 16) * DE_, 0, 0);
          acc = fmaf(aw[(size_t)b*L_ + l], er[(size_t)l * DE_], acc);
        }
        ctxb[(size_t)b*DE_ + d] = acc;
        xc16[(size_t)b*KATT + DO_ + d] = (_Float16)acc;
      }
    }
    grid_barrier(bcnt, bgen, myGen);

    // ---- B: r = relu([x,ctx] @ Wcomb^T + b_comb), stored as f16 ----------
    {
      const int NT = (B_/16) * (DH_/16);            // 128 tiles
      for (int tile = wid; tile < NT; tile += NWAVES) {
        const int mt = tile / (DH_/16), nt = tile % (DH_/16);
        const int n  = nt*16 + (lane & 15);
        v8f acc; const float bi = bcomb[n];
#pragma unroll
        for (int r = 0; r < 8; ++r) acc[r] = bi;
        acc = wmma_kloop(xc16, KATT, wcb16, KATT, mt*16, nt*16, KATT, lane, acc);
        const int mb = mt*16 + (((lane >> 4) & 1) << 3);
#pragma unroll
        for (int r = 0; r < 8; ++r)
          r16[(size_t)(mb + r) * DH_ + n] = (_Float16)fmaxf(acc[r], 0.0f);
      }
    }
    grid_barrier(bcnt, bgen, myGen);

    // ---- C: gates = r@Wih^T + h@Whh^T + biases; B operands from LDS ------
    {
      // tile == wid (512 tiles over 512 waves); this WG's slab covers its nt.
      const int rr  = blockIdx.x & 15;
      const int mt  = wid >> 7;                     // wid / 128
      const int nt  = wid & 127;
      const int n   = nt*16 + (lane & 15);
      const int nl0 = nt*16 - rr*128;               // local row base in slab
      v8f acc; const float bi = bih[n] + bhh[n];
#pragma unroll
      for (int r = 0; r < 8; ++r) acc[r] = bi;
      for (int k0 = 0; k0 < DH_; k0 += 32) {
        v16h a  = frag_a(r16, DH_, mt*16, k0, lane);
        v16h bf = frag_b(wlds, DH_, nl0, k0, lane);          // ds_load path
        acc = __builtin_amdgcn_wmma_f32_16x16x32_f16(false, a, false, bf,
                                                     (short)0, acc, false, false);
      }
      for (int k0 = 0; k0 < DH_; k0 += 32) {
        v16h a  = frag_a(h16, DH_, mt*16, k0, lane);
        v16h bf = frag_b(wlds + SLAB, DH_, nl0, k0, lane);   // ds_load path
        acc = __builtin_amdgcn_wmma_f32_16x16x32_f16(false, a, false, bf,
                                                     (short)0, acc, false, false);
      }
      store_c_f32(gates, NG, mt*16, nt*16, lane, acc);
    }
    grid_barrier(bcnt, bgen, myGen);

    // ---- C2: LSTM cell elementwise (i,f,g,o order) -----------------------
    for (int i = gtid; i < B_*DH_; i += gthr) {
      const int bb = i / DH_, j = i % DH_;
      const float* g = gates + (size_t)bb * NG;
      const float ig = g[j], fg = g[DH_ + j], gg = g[2*DH_ + j], og = g[3*DH_ + j];
      const float cn = sigf(fg) * cbuf[i] + sigf(ig) * tanhf(gg);
      const float hn = sigf(og) * tanhf(cn);
      cbuf[i] = cn; hbuf[i] = hn; h16[i] = (_Float16)hn;
    }
    grid_barrier(bcnt, bgen, myGen);

    // ---- D: out = h_new @ Wout^T + b_out (f16 result) --------------------
    {
      const int NT = (B_/16) * (DO_/16);            // 64 tiles
      for (int tile = wid; tile < NT; tile += NWAVES) {
        const int mt = tile / (DO_/16), nt = tile % (DO_/16);
        const int n  = nt*16 + (lane & 15);
        v8f acc; const float bi = bout[n];
#pragma unroll
        for (int r = 0; r < 8; ++r) acc[r] = bi;
        acc = wmma_kloop(h16, DH_, wo16, DH_, mt*16, nt*16, DH_, lane, acc);
        const int mb = mt*16 + (((lane >> 4) & 1) << 3);
#pragma unroll
        for (int r = 0; r < 8; ++r)
          o16[(size_t)(mb + r) * DO_ + n] = (_Float16)acc[r];
      }
    }
    grid_barrier(bcnt, bgen, myGen);

    // ---- E: vocab projection. Waves 0-255: common (WMMA). 256-511: batched
    if (wid < 256) {
      const int NT = (B_/16) * (CN/16);             // 256 tiles, 1 per wave
      for (int tile = wid; tile < NT; tile += 256) {
        const int mt = tile / (CN/16), nt = tile % (CN/16);
        const int n  = nt*16 + (lane & 15);
        v8f acc;
#pragma unroll
        for (int r = 0; r < 8; ++r) acc[r] = 0.0f;
        acc = wmma_kloop(o16, DO_, ce16, DO_, mt*16, nt*16, DO_, lane, acc);
        const int mb = mt*16 + (((lane >> 4) & 1) << 3);
#pragma unroll
        for (int r = 0; r < 8; ++r)
          if (n < CV_) proj[(size_t)(mb + r) * PLD + n] = acc[r];
      }
    } else {
      const int pw = wid - 256;
      for (int p = pw; p < B_*VB_; p += 256) {
        const int bb = p / VB_, v = p % VB_;
        float acc = 0.0f;
        const float* ov = obe + ((size_t)bb * VB_ + v) * DO_;
        for (int d = lane; d < DO_; d += 32)
          acc = fmaf((float)o16[(size_t)bb * DO_ + d], ov[d], acc);
#pragma unroll
        for (int o = 16; o > 0; o >>= 1) acc += __shfl_xor(acc, o, 32);
        if (lane == 0) {
          if (!obeM[(size_t)bb * VB_ + v]) acc = NEG_INF;
          proj[(size_t)bb * PLD + CV_ + v] = acc;
        }
      }
    }
    grid_barrier(bcnt, bgen, myGen);

    // ---- F: log_softmax + emit, fused with next step's input pack --------
    {
      float lm = NEG_INF;
      for (int j = threadIdx.x; j < NV_; j += NTHR)
        lm = fmaxf(lm, proj[(size_t)b*PLD + j]);
      const float mx = block_max(lm, sred);
      float ls = 0.0f;
      for (int j = threadIdx.x; j < NV_; j += NTHR)
        ls += __expf(proj[(size_t)b*PLD + j] - mx);
      const float lz = mx + __logf(block_sum(ls, sred));
      float* orow = dout + ((size_t)b * (T_-1) + t) * NV_;
      for (int j = threadIdx.x; j < NV_; j += NTHR)
        orow[j] = proj[(size_t)b*PLD + j] - lz;
      if (t + 1 < T_ - 1)
        pack_step_inputs(b, t + 1, tgt, cemb, obe, hbuf, xh16, xc16);
    }
    grid_barrier(bcnt, bgen, myGen);
  }
}

// ---------------------------------------------------------------------------
extern "C" void kernel_launch(void* const* d_in, const int* in_sizes, int n_in,
                              void* d_out, int out_size, void* d_ws, size_t ws_size,
                              hipStream_t stream) {
  (void)in_sizes; (void)n_in; (void)out_size;
  // Zero h0/c0, barrier counters, and f16 padding rows; graph-capture safe.
  size_t nbytes = WS_TOTAL <= ws_size ? WS_TOTAL : ws_size;
  (void)hipMemsetAsync(d_ws, 0, nbytes, stream);

  constexpr size_t LDS_DYN = (size_t)2 * SLAB * sizeof(_Float16);   // 256KB
  attn_lstm_decoder<<<NWG, NTHR, LDS_DYN, stream>>>(
      (const float*)d_in[0],  (const unsigned char*)d_in[1],
      (const float*)d_in[2],  (const unsigned char*)d_in[3],
      (const int*)d_in[4],    (const float*)d_in[5],
      (const float*)d_in[6],  (const float*)d_in[7],
      (const float*)d_in[8],  (const float*)d_in[9],
      (const float*)d_in[10], (const float*)d_in[11],
      (const float*)d_in[12], (const float*)d_in[13],
      (const float*)d_in[14], (const float*)d_in[15],
      (float*)d_out, (unsigned char*)d_ws);
}
